// VerticalFusion_84035330114152
// MI455X (gfx1250) — matrix operand order repaired and checked by reading, not compiled, not run on hardware
//
#include <hip/hip_runtime.h>
#include <hip/hip_bf16.h>

// ---------------- problem constants (match reference) ----------------
#define BATCH 2
#define CCH   128
#define HH1   64
#define WW1   64
#define HH2   128
#define WW2   128
#define NHD   8
#define NPT   4
#define DHEAD 16
#define NQ    (HH1*WW1)   // 4096
#define NVAL  (HH2*WW2)   // 16384
#define EPSV  1e-5f

typedef __attribute__((ext_vector_type(16))) __bf16 v16bf;
typedef __attribute__((ext_vector_type(8)))  __bf16 v8bf;
typedef __attribute__((ext_vector_type(8)))  float  v8f;

// A fragment (16x32 bf16, CDNA5 ISA §7.12.2): per lane the 16
// halves are two contiguous 8-element chunks: K = k0+8*kg+{0..7} and
// K = k0+16+8*kg+{0..7}.  Two aligned 16B loads.
__device__ __forceinline__ v16bf load_a_frag(const __bf16* __restrict__ row,
                                             int k0, int kg) {
  v8bf lo = *(const v8bf*)(row + k0 + 8 * kg);
  v8bf hi = *(const v8bf*)(row + k0 + 16 + 8 * kg);
  return __builtin_shufflevector(lo, hi, 0, 1, 2, 3, 4, 5, 6, 7,
                                 8, 9, 10, 11, 12, 13, 14, 15);
}

// B fragment (32x16 bf16): lane holds column n, halves enumerate
// K = k0+8*kg+{0..7} and K = k0+16+8*kg+{0..7} -> two aligned 16B loads.
__device__ __forceinline__ v16bf load_b_frag(const __bf16* __restrict__ wt_row,
                                             int k0, int kg) {
  v8bf lo = *(const v8bf*)(wt_row + k0 + 8 * kg);
  v8bf hi = *(const v8bf*)(wt_row + k0 + 16 + 8 * kg);
  return __builtin_shufflevector(lo, hi, 0, 1, 2, 3, 4, 5, 6, 7,
                                 8, 9, 10, 11, 12, 13, 14, 15);
}

__device__ __forceinline__ v16bf zero_frag() {
  v16bf z;
#pragma unroll
  for (int h = 0; h < 16; ++h) z[h] = (__bf16)0.f;
  return z;
}

// ======================================================================
// Generic bf16 WMMA GEMM, NT 16-wide N-tiles per wave (A reuse):
// out(MxN,f32) = A(Mx128,bf16) @ Wt(Nx128,bf16)^T + bias
// ======================================================================
template <int NT>
__global__ void k_gemm_bf16(const __bf16* __restrict__ A, const __bf16* __restrict__ Wt,
                            const float* __restrict__ bias, float* __restrict__ out,
                            int M, int N) {
  const int lane = threadIdx.x & 31;
  const int wid  = blockIdx.x * (blockDim.x >> 5) + (threadIdx.x >> 5);
  const int nGroups = N / (16 * NT);
  const int mt = wid / nGroups, ng = wid % nGroups;
  const int m0 = mt << 4, n0 = ng * 16 * NT;
  if (m0 >= M) return;
  const int row = lane & 15, kg = lane >> 4;
  const __bf16* Arow = A + (size_t)(m0 + row) * CCH;
  v8f acc[NT];
#pragma unroll
  for (int j = 0; j < NT; ++j) acc[j] = (v8f){0.f,0.f,0.f,0.f,0.f,0.f,0.f,0.f};
#pragma unroll
  for (int k0 = 0; k0 < CCH; k0 += 32) {
    const v16bf a = load_a_frag(Arow, k0, kg);
#pragma unroll
    for (int j = 0; j < NT; ++j) {
      const v16bf bm = load_b_frag(Wt + (size_t)(n0 + j * 16 + row) * CCH, k0, kg);
      acc[j] = __builtin_amdgcn_wmma_f32_16x16x32_bf16(false, a, false, bm,
                                                       (short)0, acc[j], false, false);
    }
  }
#pragma unroll
  for (int j = 0; j < NT; ++j) {
    const int nn = n0 + j * 16 + row;
    const float bv = bias ? bias[nn] : 0.f;
#pragma unroll
    for (int r = 0; r < 8; ++r)
      out[(size_t)(m0 + r + 8 * kg) * N + nn] = acc[j][r] + bv;
  }
}

// ======================================================================
// Output projection GEMM, fused +bias +identity, transposed store (B,C,NQ)
// ======================================================================
__global__ void k_out_gemm(const __bf16* __restrict__ A, const __bf16* __restrict__ Wt,
                           const float* __restrict__ bias, const float* __restrict__ ident,
                           float* __restrict__ xout) {
  constexpr int NT = 4;
  const int lane = threadIdx.x & 31;
  const int wid  = blockIdx.x * (blockDim.x >> 5) + (threadIdx.x >> 5);
  const int nGroups = CCH / (16 * NT);            // 2
  const int mt = wid / nGroups, ng = wid % nGroups;
  const int m0 = mt << 4, n0 = ng * 16 * NT;
  const int row = lane & 15, kg = lane >> 4;
  const __bf16* Arow = A + (size_t)(m0 + row) * CCH;
  v8f acc[NT];
#pragma unroll
  for (int j = 0; j < NT; ++j) acc[j] = (v8f){0.f,0.f,0.f,0.f,0.f,0.f,0.f,0.f};
#pragma unroll
  for (int k0 = 0; k0 < CCH; k0 += 32) {
    const v16bf a = load_a_frag(Arow, k0, kg);
#pragma unroll
    for (int j = 0; j < NT; ++j) {
      const v16bf bm = load_b_frag(Wt + (size_t)(n0 + j * 16 + row) * CCH, k0, kg);
      acc[j] = __builtin_amdgcn_wmma_f32_16x16x32_bf16(false, a, false, bm,
                                                       (short)0, acc[j], false, false);
    }
  }
#pragma unroll
  for (int j = 0; j < NT; ++j) {
    const int nn = n0 + j * 16 + row;
    const float bv = bias[nn];
#pragma unroll
    for (int r = 0; r < 8; ++r) {
      int m  = m0 + r + 8 * kg;                   // m = b*NQ + yx
      int b  = m >> 12;
      int yx = m & (NQ - 1);
      xout[((size_t)b * CCH + nn) * NQ + yx] = acc[j][r] + bv + ident[(size_t)m * CCH + nn];
    }
  }
}

// ======================================================================
// 3x3 conv as implicit GEMM over NHWC bf16 activations and (Cout,9,Cin)
// bf16 weights; NT channel tiles per wave, 36*NT WMMAs per wave.
// ======================================================================
__global__ void k_conv1(const __bf16* __restrict__ xn,   // (B,NVAL,128) bf16
                        const __bf16* __restrict__ wt,   // (128,9,128) bf16
                        const float* __restrict__ g, const float* __restrict__ bt,
                        const float* __restrict__ mn, const float* __restrict__ vr,
                        __bf16* __restrict__ outn) {     // (B,NVAL,128) bf16
  constexpr int NT = 4;
  const int lane = threadIdx.x & 31;
  const int wid  = blockIdx.x * (blockDim.x >> 5) + (threadIdx.x >> 5);
  const int nGroups = CCH / (16 * NT);                   // 2
  const int tilesPB = (NVAL / 16) * nGroups;             // 2048
  const int b = wid / tilesPB, t = wid % tilesPB;
  const int m0 = (t / nGroups) << 4, n0 = (t % nGroups) * 16 * NT;
  const int row = lane & 15, kg = lane >> 4;
  const int p = m0 + row, py = p >> 7, px = p & (WW2 - 1);
  const v16bf zf = zero_frag();
  v8f acc[NT];
#pragma unroll
  for (int j = 0; j < NT; ++j) acc[j] = (v8f){0.f,0.f,0.f,0.f,0.f,0.f,0.f,0.f};
#pragma unroll
  for (int t9 = 0; t9 < 9; ++t9) {
    const int yy = py + (t9 / 3) - 1;
    const int xx = px + (t9 % 3) - 1;
    const bool ok = (yy >= 0) && (yy < HH2) && (xx >= 0) && (xx < WW2);
    const __bf16* Arow = xn + ((size_t)b * NVAL + yy * WW2 + xx) * CCH;
#pragma unroll
    for (int k0 = 0; k0 < CCH; k0 += 32) {
      const v16bf a = ok ? load_a_frag(Arow, k0, kg) : zf;
#pragma unroll
      for (int j = 0; j < NT; ++j) {
        const v16bf bm = load_b_frag(
            wt + ((size_t)(n0 + j * 16 + row) * 9 + t9) * CCH, k0, kg);
        acc[j] = __builtin_amdgcn_wmma_f32_16x16x32_bf16(false, a, false, bm,
                                                         (short)0, acc[j], false, false);
      }
    }
  }
#pragma unroll
  for (int j = 0; j < NT; ++j) {
    const int nn = n0 + j * 16 + row;
    const float sc = g[nn] * rsqrtf(vr[nn] + EPSV);
    const float sh = bt[nn] - mn[nn] * sc;
#pragma unroll
    for (int r = 0; r < 8; ++r) {
      int pp = m0 + r + 8 * kg;
      outn[((size_t)b * NVAL + pp) * CCH + nn] = (__bf16)fmaxf(acc[j][r] * sc + sh, 0.f);
    }
  }
}

__global__ void k_conv2(const __bf16* __restrict__ kn,   // (B,NVAL,128) bf16
                        const __bf16* __restrict__ wt,   // (25,9,128) bf16
                        const float* __restrict__ g, const float* __restrict__ bt,
                        const float* __restrict__ mn, const float* __restrict__ vr,
                        float* __restrict__ ker) {       // (B,25,NVAL) f32
  constexpr int NT = 2;                                  // 32 cols, 25 valid
  const int lane = threadIdx.x & 31;
  const int wid  = blockIdx.x * (blockDim.x >> 5) + (threadIdx.x >> 5);
  const int tilesPB = NVAL / 16;                         // 1024 (one N group)
  const int b = wid / tilesPB;
  const int m0 = (wid % tilesPB) << 4;
  const int row = lane & 15, kg = lane >> 4;
  const int p = m0 + row, py = p >> 7, px = p & (WW2 - 1);
  const v16bf zf = zero_frag();
  v8f acc[NT];
#pragma unroll
  for (int j = 0; j < NT; ++j) acc[j] = (v8f){0.f,0.f,0.f,0.f,0.f,0.f,0.f,0.f};
#pragma unroll
  for (int t9 = 0; t9 < 9; ++t9) {
    const int yy = py + (t9 / 3) - 1;
    const int xx = px + (t9 % 3) - 1;
    const bool ok = (yy >= 0) && (yy < HH2) && (xx >= 0) && (xx < WW2);
    const __bf16* Arow = kn + ((size_t)b * NVAL + yy * WW2 + xx) * CCH;
#pragma unroll
    for (int k0 = 0; k0 < CCH; k0 += 32) {
      const v16bf a = ok ? load_a_frag(Arow, k0, kg) : zf;
#pragma unroll
      for (int j = 0; j < NT; ++j) {
        const int co = j * 16 + row;
        const v16bf bm = (co < 25)
            ? load_b_frag(wt + ((size_t)co * 9 + t9) * CCH, k0, kg) : zf;
        acc[j] = __builtin_amdgcn_wmma_f32_16x16x32_bf16(false, a, false, bm,
                                                         (short)0, acc[j], false, false);
      }
    }
  }
#pragma unroll
  for (int j = 0; j < NT; ++j) {
    const int nn = j * 16 + row;
    if (nn >= 25) continue;
    const float sc = g[nn] * rsqrtf(vr[nn] + EPSV);
    const float sh = bt[nn] - mn[nn] * sc;
#pragma unroll
    for (int r = 0; r < 8; ++r) {
      int pp = m0 + r + 8 * kg;
      ker[((size_t)b * 25 + nn) * NVAL + pp] = fmaxf(acc[j][r] * sc + sh, 0.f);
    }
  }
}

// ======================================================================
// Prep kernels: weights -> bf16 K-contiguous, x2 -> NHWC bf16
// ======================================================================
__global__ void k_transpose_w(const float* __restrict__ w,   // (128,N)
                              __bf16* __restrict__ wt, int N) { // (N,128)
  int i = blockIdx.x * blockDim.x + threadIdx.x;
  if (i >= N * CCH) return;
  int n = i >> 7, k = i & (CCH - 1);
  wt[i] = (__bf16)w[(size_t)k * N + n];
}

__global__ void k_prep_conv_w(const float* __restrict__ w,   // (Cout,128,3,3)
                              __bf16* __restrict__ wt, int total) { // (Cout,9,128)
  int i = blockIdx.x * blockDim.x + threadIdx.x;
  if (i >= total) return;
  int co = i / 1152, r = i % 1152;
  int t = r >> 7, ci = r & (CCH - 1);
  wt[i] = (__bf16)w[((size_t)co * CCH + ci) * 9 + t];
}

__global__ void k_nhwc_bf16(const float* __restrict__ x,     // (B,C,NVAL)
                            __bf16* __restrict__ xn) {       // (B,NVAL,C)
  int i = blockIdx.x * blockDim.x + threadIdx.x;
  if (i >= BATCH * NVAL * CCH) return;
  int c = i & (CCH - 1), p = (i >> 7) & (NVAL - 1), b = i >> 21;
  xn[i] = (__bf16)x[((size_t)b * CCH + c) * NVAL + p];
}

// ======================================================================
// Elementwise / normalization kernels
// ======================================================================
__global__ void k_ln2(const float* __restrict__ x2, const float* __restrict__ g,
                      const float* __restrict__ be, __bf16* __restrict__ out) {
  int tid = blockIdx.x * blockDim.x + threadIdx.x;
  if (tid >= BATCH * NVAL) return;
  int b = tid >> 14, n = tid & (NVAL - 1);
  const float* base = x2 + (size_t)b * CCH * NVAL + n;
  float s = 0.f;
  for (int c = 0; c < CCH; ++c) s += base[(size_t)c * NVAL];
  float mean = s * (1.f / CCH);
  float v = 0.f;
  for (int c = 0; c < CCH; ++c) { float d = base[(size_t)c * NVAL] - mean; v += d * d; }
  float inv = rsqrtf(v * (1.f / CCH) + EPSV);
  for (int c = 0; c < CCH; ++c)
    out[(size_t)tid * CCH + c] = (__bf16)((base[(size_t)c * NVAL] - mean) * inv * g[c] + be[c]);
}

__global__ void k_ln1_pos(const float* __restrict__ x1, const float* __restrict__ g,
                          const float* __restrict__ be, const float* __restrict__ ps,
                          float* __restrict__ qid, __bf16* __restrict__ qpb) {
  int tid = blockIdx.x * blockDim.x + threadIdx.x;
  if (tid >= BATCH * NQ) return;
  int b = tid >> 12, n = tid & (NQ - 1);
  const float* base = x1 + (size_t)b * CCH * NQ + n;
  float s = 0.f;
  for (int c = 0; c < CCH; ++c) s += base[(size_t)c * NQ];
  float mean = s * (1.f / CCH);
  float v = 0.f;
  for (int c = 0; c < CCH; ++c) { float d = base[(size_t)c * NQ] - mean; v += d * d; }
  float inv = rsqrtf(v * (1.f / CCH) + EPSV);
  float scale = ps[0];
  float tn = (float)n;
  for (int c = 0; c < CCH; ++c) {
    float q = (base[(size_t)c * NQ] - mean) * inv * g[c] + be[c];
    qid[(size_t)tid * CCH + c] = q;
    int i = (c < 64) ? c : (c - 64);
    float freq = __expf(-0.14391157f * (float)i);      // 1/10000^(2i/128)
    float pe = ((c < 64) ? __sinf(tn * freq) : __cosf(tn * freq)) * scale;
    qpb[(size_t)tid * CCH + c] = (__bf16)(q + pe);
  }
}

__global__ void k_softmax_aw(float* __restrict__ aw) {       // (B,NQ,NH,4) in place
  int tid = blockIdx.x * blockDim.x + threadIdx.x;
  if (tid >= BATCH * NQ * NHD) return;
  float* p = aw + (size_t)tid * NPT;
  float m = fmaxf(fmaxf(p[0], p[1]), fmaxf(p[2], p[3]));
  float e0 = __expf(p[0] - m), e1 = __expf(p[1] - m);
  float e2 = __expf(p[2] - m), e3 = __expf(p[3] - m);
  float inv = 1.f / (e0 + e1 + e2 + e3);
  p[0] = e0 * inv; p[1] = e1 * inv; p[2] = e2 * inv; p[3] = e3 * inv;
}

__global__ void k_sample_attn(const float* __restrict__ off, const float* __restrict__ aw,
                              const float* __restrict__ val, __bf16* __restrict__ attnb) {
  int tid = blockIdx.x * blockDim.x + threadIdx.x;
  if (tid >= BATCH * NQ * NHD) return;
  int h  = tid & 7;
  int bq = tid >> 3;
  int q  = bq & (NQ - 1);
  int b  = bq >> 12;
  int iy = q >> 6, ix = q & (WW1 - 1);
  float acc[DHEAD];
#pragma unroll
  for (int d = 0; d < DHEAD; ++d) acc[d] = 0.f;
  const float* offp = off + (size_t)bq * 64 + h * 8;
  const float* awp  = aw  + (size_t)bq * 32 + h * 4;
  for (int p = 0; p < NPT; ++p) {
    float sx = 2.f * ix + 0.5f + offp[p * 2 + 0];      // grid_sample collapsed
    float sy = 2.f * iy + 0.5f + offp[p * 2 + 1];
    float fx = floorf(sx), fy = floorf(sy);
    float wx = sx - fx, wy = sy - fy;
    int x0 = (int)fx, y0 = (int)fy;
    float wpt = awp[p];
#pragma unroll
    for (int cr = 0; cr < 4; ++cr) {
      int xi = x0 + (cr & 1), yi = y0 + (cr >> 1);
      if (xi < 0 || xi >= WW2 || yi < 0 || yi >= HH2) continue;
      float wgt = ((cr & 1) ? wx : 1.f - wx) * ((cr >> 1) ? wy : 1.f - wy) * wpt;
      const float* vp = val + ((size_t)b * NVAL + yi * WW2 + xi) * CCH + h * DHEAD;
#pragma unroll
      for (int d = 0; d < DHEAD; ++d) acc[d] += wgt * vp[d];
    }
  }
  __bf16* o = attnb + (size_t)bq * CCH + h * DHEAD;
#pragma unroll
  for (int d = 0; d < DHEAD; ++d) o[d] = (__bf16)acc[d];
}

__global__ void k_softmax25(float* __restrict__ ker) {
  int tid = blockIdx.x * blockDim.x + threadIdx.x;
  if (tid >= BATCH * NVAL) return;
  int b = tid >> 14, n = tid & (NVAL - 1);
  float* p = ker + (size_t)b * 25 * NVAL + n;
  float v[25], m = -1e30f;
#pragma unroll
  for (int t = 0; t < 25; ++t) { v[t] = p[(size_t)t * NVAL]; m = fmaxf(m, v[t]); }
  float s = 0.f;
#pragma unroll
  for (int t = 0; t < 25; ++t) { v[t] = __expf(v[t] - m); s += v[t]; }
  float inv = 1.f / s;
#pragma unroll
  for (int t = 0; t < 25; ++t) p[(size_t)t * NVAL] = v[t] * inv;
}

__global__ void k_fuse(const float* __restrict__ x2, const float* __restrict__ xq,
                       const float* __restrict__ ker, float* __restrict__ out) {
  int i = blockIdx.x * blockDim.x + threadIdx.x;       // b*C*NVAL + c*NVAL + pix
  if (i >= BATCH * CCH * NVAL) return;
  int pix = i & (NVAL - 1);
  int c   = (i >> 14) & (CCH - 1);
  int b   = i >> 21;
  int y = pix >> 7, x = pix & (WW2 - 1);
  float s = x2[i];
#pragma unroll
  for (int t = 0; t < 25; ++t) {
    int yy = y + 2 * (t / 5) - 4;
    int xx = x + 2 * (t % 5) - 4;
    if (yy < 0 || yy >= HH2 || xx < 0 || xx >= WW2) continue;
    float kv = ker[((size_t)b * 25 + t) * NVAL + pix];
    float xv = xq[((size_t)b * CCH + c) * NQ + (yy >> 1) * WW1 + (xx >> 1)];
    s += kv * xv;
  }
  out[i] = s;
}

// ======================================================================
extern "C" void kernel_launch(void* const* d_in, const int* in_sizes, int n_in,
                              void* d_out, int out_size, void* d_ws, size_t ws_size,
                              hipStream_t stream) {
  const float* x1     = (const float*)d_in[0];
  const float* x2     = (const float*)d_in[1];
  const float* ln1_g  = (const float*)d_in[2];
  const float* ln1_b  = (const float*)d_in[3];
  const float* ln2_g  = (const float*)d_in[4];
  const float* ln2_b  = (const float*)d_in[5];
  const float* pos_s  = (const float*)d_in[6];
  const float* off_w  = (const float*)d_in[7];
  const float* off_b  = (const float*)d_in[8];
  const float* aw_w   = (const float*)d_in[9];
  const float* aw_b   = (const float*)d_in[10];
  const float* vp_w   = (const float*)d_in[11];
  const float* vp_b   = (const float*)d_in[12];
  const float* op_w   = (const float*)d_in[13];
  const float* op_b   = (const float*)d_in[14];
  const float* conv_w = (const float*)d_in[15];
  const float* cbn_g  = (const float*)d_in[16];
  const float* cbn_b  = (const float*)d_in[17];
  const float* cbn_m  = (const float*)d_in[18];
  const float* cbn_v  = (const float*)d_in[19];
  const float* enc_w  = (const float*)d_in[20];
  const float* ebn_g  = (const float*)d_in[21];
  const float* ebn_b  = (const float*)d_in[22];
  const float* ebn_m  = (const float*)d_in[23];
  const float* ebn_v  = (const float*)d_in[24];
  float* out = (float*)d_out;

  char* ws = (char*)d_ws;
  size_t o = 0;
  auto take = [&](size_t bytes) -> void* {
    void* r = ws + o; o += (bytes + 255) & ~(size_t)255; return r;
  };
  __bf16* x2n   = (__bf16*)take((size_t)BATCH * NVAL * CCH * 2);  // x2 NHWC bf16
  __bf16* ln2b  = (__bf16*)take((size_t)BATCH * NVAL * CCH * 2);  // LN(x2) bf16
  float*  val   = (float*) take((size_t)BATCH * NVAL * CCH * 4);  // value proj f32
  float*  qid   = (float*) take((size_t)BATCH * NQ * CCH * 4);    // identity q
  __bf16* qpb   = (__bf16*)take((size_t)BATCH * NQ * CCH * 2);    // q + pos bf16
  float*  offb  = (float*) take((size_t)BATCH * NQ * 64 * 4);
  float*  awb   = (float*) take((size_t)BATCH * NQ * 32 * 4);
  __bf16* attnb = (__bf16*)take((size_t)BATCH * NQ * CCH * 2);
  float*  xproj = (float*) take((size_t)BATCH * CCH * NQ * 4);    // x (B,C,64,64)
  __bf16* kn    = (__bf16*)take((size_t)BATCH * NVAL * CCH * 2);  // conv1 act NHWC
  float*  ker   = (float*) take((size_t)BATCH * 25 * NVAL * 4);
  __bf16* vpwt  = (__bf16*)take((size_t)CCH * CCH * 2);           // (128,128)
  __bf16* opwt  = (__bf16*)take((size_t)CCH * CCH * 2);
  __bf16* offwt = (__bf16*)take((size_t)64 * CCH * 2);
  __bf16* awwt  = (__bf16*)take((size_t)32 * CCH * 2);
  __bf16* cwt   = (__bf16*)take((size_t)CCH * 9 * CCH * 2);       // (128,9,128)
  __bf16* ewt   = (__bf16*)take((size_t)25 * 9 * CCH * 2);        // (25,9,128)

  const int T = 256;
  // --- weight / layout prep (tiny, one-shot per call) ---
  k_transpose_w<<<(CCH * CCH + T - 1) / T, T, 0, stream>>>(vp_w, vpwt, CCH);
  k_transpose_w<<<(CCH * CCH + T - 1) / T, T, 0, stream>>>(op_w, opwt, CCH);
  k_transpose_w<<<(64 * CCH + T - 1) / T, T, 0, stream>>>(off_w, offwt, 64);
  k_transpose_w<<<(32 * CCH + T - 1) / T, T, 0, stream>>>(aw_w, awwt, 32);
  k_prep_conv_w<<<(CCH * 1152 + T - 1) / T, T, 0, stream>>>(conv_w, cwt, CCH * 1152);
  k_prep_conv_w<<<(25 * 1152 + T - 1) / T, T, 0, stream>>>(enc_w, ewt, 25 * 1152);
  k_nhwc_bf16<<<(BATCH * NVAL * CCH + T - 1) / T, T, 0, stream>>>(x2, x2n);
  // --- value branch ---
  k_ln2<<<(BATCH * NVAL + T - 1) / T, T, 0, stream>>>(x2, ln2_g, ln2_b, ln2b);
  k_gemm_bf16<4><<<(BATCH * NVAL / 16) * 2 / 8, T, 0, stream>>>(
      ln2b, vpwt, vp_b, val, BATCH * NVAL, CCH);
  // --- query branch ---
  k_ln1_pos<<<(BATCH * NQ + T - 1) / T, T, 0, stream>>>(x1, ln1_g, ln1_b, pos_s, qid, qpb);
  k_gemm_bf16<4><<<(BATCH * NQ / 16) * 1 / 8, T, 0, stream>>>(
      qpb, offwt, off_b, offb, BATCH * NQ, 64);
  k_gemm_bf16<2><<<(BATCH * NQ / 16) * 1 / 8, T, 0, stream>>>(
      qpb, awwt, aw_b, awb, BATCH * NQ, 32);
  k_softmax_aw<<<(BATCH * NQ * NHD + T - 1) / T, T, 0, stream>>>(awb);
  // --- deformable sampling + output projection ---
  k_sample_attn<<<(BATCH * NQ * NHD + T - 1) / T, T, 0, stream>>>(offb, awb, val, attnb);
  k_out_gemm<<<(BATCH * NQ / 16) * 2 / 8, T, 0, stream>>>(attnb, opwt, op_b, qid, xproj);
  // --- CARAFE kernel prediction (two 3x3 convs via WMMA implicit GEMM) ---
  k_conv1<<<BATCH * (NVAL / 16) * 2 / 8, T, 0, stream>>>(
      x2n, cwt, cbn_g, cbn_b, cbn_m, cbn_v, kn);
  k_conv2<<<BATCH * (NVAL / 16) / 8, T, 0, stream>>>(
      kn, ewt, ebn_g, ebn_b, ebn_m, ebn_v, ker);
  k_softmax25<<<(BATCH * NVAL + T - 1) / T, T, 0, stream>>>(ker);
  // --- fuse ---
  k_fuse<<<(BATCH * CCH * NVAL + T - 1) / T, T, 0, stream>>>(x2, xproj, ker, out);
}
